// W4A8QGLA_5909874999712
// MI455X (gfx1250) — 8-bit matrix operand order repaired and checked by reading, compiled, not run on hardware
//
#include <hip/hip_runtime.h>
#include <hip/hip_bf16.h>
#include <math.h>

#define BB 2
#define TT 4096
#define DD 2048
#define HH 16
#define DKK 128
#define DVV 128
#define MM (BB * TT)
#define KD DD
#define ND DD
#define GATE_NORM_INV (1.0f / 16.0f)
#define RMS_EPS 1e-5f
#define QK_SCALE 0.08838834764831845f

typedef __attribute__((ext_vector_type(8))) int v8i;

__global__ void qgla_quant_pack_a(const float* __restrict__ X,
                                  const float* __restrict__ ascale_p,
                                  int* __restrict__ Aq, int M, int K) {
    const int KT = K >> 6;
    const size_t tid = (size_t)blockIdx.x * blockDim.x + threadIdx.x;
    const size_t total = (size_t)M * K / 4;
    if (tid >= total) return;
    const int tile = (int)(tid >> 8);
    const int r    = (int)(tid & 255);
    const int lane = r >> 3;
    const int j    = r & 7;
    const int mt = tile / KT, kt = tile % KT;
    const int m  = mt * 16 + (lane & 15);
    const int kb = kt * 64 + ((j >> 1) * 16) + ((j & 1) * 4) + ((lane >> 4) * 8);
    const float inv = 1.0f / ascale_p[0];
    const float* xr = X + (size_t)m * K + kb;
    int packed = 0;
#pragma unroll
    for (int b = 0; b < 4; ++b) {
        float q = rintf(xr[b] * inv);
        q = fminf(fmaxf(q, -128.0f), 127.0f);
        packed |= ((int)q & 0xff) << (8 * b);
    }
    Aq[tid] = packed;
}

__global__ void qgla_pack_w(const int* __restrict__ W,
                            int* __restrict__ Wpk, int N, int K) {
    const int KT = K >> 6;
    const size_t tid = (size_t)blockIdx.x * blockDim.x + threadIdx.x;
    const size_t total = (size_t)N * K / 4;
    if (tid >= total) return;
    const int tile = (int)(tid >> 8);
    const int r    = (int)(tid & 255);
    const int lane = r >> 3;
    const int j    = r & 7;
    const int nt = tile / KT, kt = tile % KT;
    const int n  = nt * 16 + (lane & 15);
    const int kb = kt * 64 + ((j >> 1) * 16) + ((j & 1) * 4) + ((lane >> 4) * 8);
    const int* wr = W + (size_t)n * K + kb;
    int packed = 0;
#pragma unroll
    for (int b = 0; b < 4; ++b) packed |= (wr[b] & 0xff) << (8 * b);
    Wpk[tid] = packed;
}

__global__ void __launch_bounds__(256)
qgla_gemm_iu8(const int* __restrict__ Aq, const int* __restrict__ Wpk,
              const float* __restrict__ wscale, const float* __restrict__ ascale_p,
              float* __restrict__ C, int M, int N, int K) {
    const int KT   = K >> 6;
    const int lane = threadIdx.x & 31;
    const int w    = threadIdx.x >> 5;
    const int wm   = w & 1;
    const int wn   = w >> 1;
    const int mt0  = blockIdx.x * 4 + wm * 2;
    const int nt0  = blockIdx.y * 8 + wn * 2;
    const int laneOff = lane * 8;

    v8i acc00 = {}, acc01 = {}, acc10 = {}, acc11 = {};

    const int* ap0 = Aq  + (size_t)(mt0 * KT) * 256 + laneOff;
    const int* ap1 = Aq  + (size_t)((mt0 + 1) * KT) * 256 + laneOff;
    const int* bp0 = Wpk + (size_t)(nt0 * KT) * 256 + laneOff;
    const int* bp1 = Wpk + (size_t)((nt0 + 1) * KT) * 256 + laneOff;

    for (int kt = 0; kt < KT; ++kt) {
        const v8i a0 = *(const v8i*)(ap0 + (size_t)kt * 256);
        const v8i a1 = *(const v8i*)(ap1 + (size_t)kt * 256);
        const v8i b0 = *(const v8i*)(bp0 + (size_t)kt * 256);
        const v8i b1 = *(const v8i*)(bp1 + (size_t)kt * 256);
        if (kt + 1 < KT) {
            __builtin_prefetch(ap0 + (size_t)(kt + 1) * 256, 0, 1);
            __builtin_prefetch(bp0 + (size_t)(kt + 1) * 256, 0, 1);
        }
        acc00 = __builtin_amdgcn_wmma_i32_16x16x64_iu8(true, a0, true, b0, acc00, false, false);
        acc01 = __builtin_amdgcn_wmma_i32_16x16x64_iu8(true, a0, true, b1, acc01, false, false);
        acc10 = __builtin_amdgcn_wmma_i32_16x16x64_iu8(true, a1, true, b0, acc10, false, false);
        acc11 = __builtin_amdgcn_wmma_i32_16x16x64_iu8(true, a1, true, b1, acc11, false, false);
    }

    const float asc  = ascale_p[0];
    const int   nl   = lane & 15;
    const int   mhi  = (lane >> 4) * 8;

    auto store_tile = [&](const v8i& acc, int mt, int nt) {
        const int n = nt * 16 + nl;
        const float sc = asc * wscale[n];
#pragma unroll
        for (int r = 0; r < 8; ++r) {
            const int m = mt * 16 + mhi + r;
            C[(size_t)m * N + n] = (float)acc[r] * sc;
        }
    };
    store_tile(acc00, mt0,     nt0);
    store_tile(acc01, mt0,     nt0 + 1);
    store_tile(acc10, mt0 + 1, nt0);
    store_tile(acc11, mt0 + 1, nt0 + 1);
}

__global__ void qgla_gk0(const float* __restrict__ X, const int* __restrict__ W,
                         const float* __restrict__ wscale, const float* __restrict__ ascale_p,
                         float* __restrict__ out) {
    const size_t tid = (size_t)blockIdx.x * blockDim.x + threadIdx.x;
    if (tid >= (size_t)MM * 16) return;
    const int m = (int)(tid >> 4);
    const int n = (int)(tid & 15);
    const float as  = ascale_p[0];
    const float inv = 1.0f / as;
    const float* xr = X + (size_t)m * KD;
    const int*   wr = W + (size_t)n * KD;
    int acc = 0;
    for (int k = 0; k < KD; ++k) {
        float q = rintf(xr[k] * inv);
        q = fminf(fmaxf(q, -128.0f), 127.0f);
        acc += (int)q * wr[k];
    }
    out[tid] = (float)acc * as * wscale[n];
}

__global__ void qgla_gk1(const float* __restrict__ X16, const int* __restrict__ W,
                         const float* __restrict__ wscale, const float* __restrict__ ascale_p,
                         const float* __restrict__ bias, float* __restrict__ gk) {
    const size_t tid = (size_t)blockIdx.x * blockDim.x + threadIdx.x;
    if (tid >= (size_t)MM * ND) return;
    const int m = (int)(tid / ND);
    const int n = (int)(tid % ND);
    const float as  = ascale_p[0];
    const float inv = 1.0f / as;
    const float* xr = X16 + (size_t)m * 16;
    const int*   wr = W + (size_t)n * 16;
    int acc = 0;
#pragma unroll
    for (int j = 0; j < 16; ++j) {
        float q = rintf(xr[j] * inv);
        q = fminf(fmaxf(q, -128.0f), 127.0f);
        acc += (int)q * wr[j];
    }
    const float x = (float)acc * as * wscale[n] + bias[n];
    const float ls = fminf(x, 0.0f) - log1pf(expf(-fabsf(x)));
    gk[tid] = ls * GATE_NORM_INV;
}

__global__ void __launch_bounds__(256)
qgla_scan(const float* __restrict__ q, const float* __restrict__ k,
          const float* __restrict__ v, const float* __restrict__ gk,
          float* __restrict__ o) {
    __shared__ float qs[DKK], ks[DKK], egk[DKK], vs[DVV], part[256];
    const int bh = blockIdx.x;
    const int b  = bh / HH;
    const int h  = bh % HH;
    const int t  = threadIdx.x;
    const int vc = t & 127;
    const int kb = (t >> 7) * 64;

    float S[64];
#pragma unroll
    for (int i = 0; i < 64; ++i) S[i] = 0.0f;

    const size_t base0 = (size_t)b * TT * DD + (size_t)h * DKK;
    for (int tt = 0; tt < TT; ++tt) {
        const size_t base = base0 + (size_t)tt * DD;
        if (t < 128) {
            qs[t]  = q[base + t] * QK_SCALE;
            ks[t]  = k[base + t];
            egk[t] = expf(gk[base + t]);
        } else {
            vs[t - 128] = v[base + (t - 128)];
        }
        __syncthreads();
        const float vv = vs[vc];
        float acc = 0.0f;
#pragma unroll
        for (int kk = 0; kk < 64; ++kk) {
            float s = S[kk];
            s = egk[kb + kk] * s + ks[kb + kk] * vv;
            S[kk] = s;
            acc += qs[kb + kk] * s;
        }
        part[t] = acc;
        __syncthreads();
        if (t < 128) o[base + t] = part[t] + part[t + 128];
        __syncthreads();
    }
}

__global__ void __launch_bounds__(128)
qgla_norm_gate(const float* __restrict__ o_in, const float* __restrict__ g,
               const float* __restrict__ norm_w, float* __restrict__ out) {
    __shared__ float red[128];
    const size_t base = (size_t)blockIdx.x * DVV;
    const int t = threadIdx.x;
    const float ov = o_in[base + t];
    red[t] = ov * ov;
    __syncthreads();
#pragma unroll
    for (int s = 64; s > 0; s >>= 1) {
        if (t < s) red[t] += red[t + s];
        __syncthreads();
    }
    const float r  = rsqrtf(red[0] * (1.0f / DVV) + RMS_EPS);
    const float gv = g[base + t];
    const float sw = gv / (1.0f + expf(-gv));
    out[base + t] = ov * r * norm_w[t] * sw;
}

extern "C" void kernel_launch(void* const* d_in, const int* in_sizes, int n_in,
                              void* d_out, int out_size, void* d_ws, size_t ws_size,
                              hipStream_t stream) {
    const float* x       = (const float*)d_in[0];
    const int*   q_w     = (const int*)d_in[1];
    const float* q_ws    = (const float*)d_in[2];
    const float* q_as    = (const float*)d_in[3];
    const int*   k_w     = (const int*)d_in[4];
    const float* k_ws    = (const float*)d_in[5];
    const float* k_as    = (const float*)d_in[6];
    const int*   v_w     = (const int*)d_in[7];
    const float* v_ws    = (const float*)d_in[8];
    const float* v_as    = (const float*)d_in[9];
    const int*   g_w     = (const int*)d_in[10];
    const float* g_ws    = (const float*)d_in[11];
    const float* g_as    = (const float*)d_in[12];
    const int*   o_w     = (const int*)d_in[13];
    const float* o_ws    = (const float*)d_in[14];
    const float* o_as    = (const float*)d_in[15];
    const int*   gk0_w   = (const int*)d_in[16];
    const float* gk0_ws  = (const float*)d_in[17];
    const float* gk0_as  = (const float*)d_in[18];
    const int*   gk1_w   = (const int*)d_in[19];
    const float* gk1_ws  = (const float*)d_in[20];
    const float* gk1_as  = (const float*)d_in[21];
    const float* gk1_b   = (const float*)d_in[22];
    const float* gnw     = (const float*)d_in[23];
    float* out = (float*)d_out;

    char* ws = (char*)d_ws;
    const size_t AQ_B   = (size_t)MM * KD;
    const size_t WPK_B  = (size_t)ND * KD;
    const size_t F32_B  = (size_t)MM * ND * 4;
    size_t off = 0;
    int* aq    = (int*)(ws + off); off += AQ_B;
    int* wq_q  = (int*)(ws + off); off += WPK_B;
    int* wq_k  = (int*)(ws + off); off += WPK_B;
    int* wq_v  = (int*)(ws + off); off += WPK_B;
    int* wq_g  = (int*)(ws + off); off += WPK_B;
    int* wq_o  = (int*)(ws + off); off += WPK_B;
    float* qf     = (float*)(ws + off); off += F32_B;
    float* kf     = (float*)(ws + off); off += F32_B;
    float* vf     = (float*)(ws + off); off += F32_B;
    float* gf     = (float*)(ws + off); off += F32_B;
    float* gkf    = (float*)(ws + off); off += F32_B;
    float* o_scan = (float*)(ws + off); off += F32_B;
    float* gk0out = (float*)(ws + off); off += (size_t)MM * 16 * 4;

    const dim3 gemm_grid(MM / 64, ND / 128);
    const int  pack_a_blocks = (int)(((size_t)MM * KD / 4 + 255) / 256);
    const int  pack_w_blocks = (int)(((size_t)ND * KD / 4 + 255) / 256);

    qgla_pack_w<<<pack_w_blocks, 256, 0, stream>>>(q_w, wq_q, ND, KD);
    qgla_pack_w<<<pack_w_blocks, 256, 0, stream>>>(k_w, wq_k, ND, KD);
    qgla_pack_w<<<pack_w_blocks, 256, 0, stream>>>(v_w, wq_v, ND, KD);
    qgla_pack_w<<<pack_w_blocks, 256, 0, stream>>>(g_w, wq_g, ND, KD);
    qgla_pack_w<<<pack_w_blocks, 256, 0, stream>>>(o_w, wq_o, ND, KD);

    qgla_quant_pack_a<<<pack_a_blocks, 256, 0, stream>>>(x, q_as, aq, MM, KD);
    qgla_gemm_iu8<<<gemm_grid, 256, 0, stream>>>(aq, wq_q, q_ws, q_as, qf, MM, ND, KD);
    qgla_quant_pack_a<<<pack_a_blocks, 256, 0, stream>>>(x, k_as, aq, MM, KD);
    qgla_gemm_iu8<<<gemm_grid, 256, 0, stream>>>(aq, wq_k, k_ws, k_as, kf, MM, ND, KD);
    qgla_quant_pack_a<<<pack_a_blocks, 256, 0, stream>>>(x, v_as, aq, MM, KD);
    qgla_gemm_iu8<<<gemm_grid, 256, 0, stream>>>(aq, wq_v, v_ws, v_as, vf, MM, ND, KD);
    qgla_quant_pack_a<<<pack_a_blocks, 256, 0, stream>>>(x, g_as, aq, MM, KD);
    qgla_gemm_iu8<<<gemm_grid, 256, 0, stream>>>(aq, wq_g, g_ws, g_as, gf, MM, ND, KD);

    qgla_gk0<<<(MM * 16 + 255) / 256, 256, 0, stream>>>(x, gk0_w, gk0_ws, gk0_as, gk0out);
    qgla_gk1<<<(int)(((size_t)MM * ND + 255) / 256), 256, 0, stream>>>(
        gk0out, gk1_w, gk1_ws, gk1_as, gk1_b, gkf);

    qgla_scan<<<BB * HH, 256, 0, stream>>>(qf, kf, vf, gkf, o_scan);

    qgla_norm_gate<<<MM * HH, 128, 0, stream>>>(o_scan, gf, gnw, qf);

    qgla_quant_pack_a<<<pack_a_blocks, 256, 0, stream>>>(qf, o_as, aq, MM, KD);
    qgla_gemm_iu8<<<gemm_grid, 256, 0, stream>>>(aq, wq_o, o_ws, o_as, out, MM, ND, KD);
}
